// RNNWavefunction1D_12025908429427
// MI455X (gfx1250) — matrix operand order repaired and checked by reading, not compiled, not run on hardware
//
#include <hip/hip_runtime.h>
#include <math.h>

// Problem constants (from reference)
#define NSITES 256
#define NHID   64
#define KDIM   2
#define BATCH  8192

#define WAVES          8
#define ROWS_PER_WAVE  16
#define BLOCK_ROWS     (WAVES * ROWS_PER_WAVE)   // 128 samples / block
#define HPITCH         68                        // f32 pitch for h staging (272B, 16B aligned, conflict-free)

typedef _Float16 v16h __attribute__((ext_vector_type(16)));
typedef float    v8f  __attribute__((ext_vector_type(8)));

__device__ __forceinline__ float sigmoidf_(float x) {
    return 1.0f / (1.0f + __expf(-x));
}

// splitmix64-style hash -> uniform [0,1), deterministic in (step, row)
__device__ __forceinline__ float uhash(unsigned step, unsigned row) {
    unsigned long long x = ((unsigned long long)row << 32) | (unsigned long long)(step + 0x9E3779B9u);
    x ^= x >> 30; x *= 0xBF58476D1CE4E5B9ULL;
    x ^= x >> 27; x *= 0x94D049BB133111EBULL;
    x ^= x >> 31;
    return (float)(unsigned)(x >> 40) * (1.0f / 16777216.0f);
}

__global__ __launch_bounds__(256) void rnn_wavefn_sample_kernel(
    const float* __restrict__ Wi,   // [K, 3*NH] = [2,192]
    const float* __restrict__ Wh,   // [NH, 3*NH] = [64,192]
    const float* __restrict__ b,    // [2, 3*NH]
    const float* __restrict__ Wd,   // [NH, K] = [64,2]
    const float* __restrict__ bd,   // [K]
    float* __restrict__ out)        // samples [B*N] then log_probs [B]
{
    // LDS: Wh transposed as f16 (B-fragment friendly), mx table, biases, Wd, h staging
    __shared__ _Float16 whT[192 * 64];                 // whT[n*64 + k] = Wh[k][n]   (24576 B)
    __shared__ float    mxTab[3 * 192];                // row0=b0, row1=b0+Wi[0], row2=b0+Wi[1]
    __shared__ float    b1s[192];
    __shared__ float    WdL[NHID * KDIM];
    __shared__ float    bdL[KDIM];
    __shared__ float    hbuf[WAVES][ROWS_PER_WAVE * HPITCH];   // per-wave h staging (34816 B)

    const int tid = threadIdx.x;

    // ---- stage weights into LDS ----
    for (int i = tid; i < NHID * 192; i += blockDim.x) {
        int k = i / 192, n = i % 192;
        whT[n * 64 + k] = (_Float16)Wh[i];
    }
    for (int j = tid; j < 192; j += blockDim.x) {
        float b0 = b[j];
        mxTab[0 * 192 + j] = b0;
        mxTab[1 * 192 + j] = b0 + Wi[0 * 192 + j];
        mxTab[2 * 192 + j] = b0 + Wi[1 * 192 + j];
        b1s[j] = b[192 + j];
    }
    for (int i = tid; i < NHID * KDIM; i += blockDim.x) WdL[i] = Wd[i];
    if (tid < KDIM) bdL[tid] = bd[tid];
    __syncthreads();

    const int wave = tid >> 5;
    const int lane = tid & 31;
    const int ln   = lane & 15;   // 0..15
    const int lh   = lane >> 4;   // 0 or 1
    const int rowBase = blockIdx.x * BLOCK_ROWS + wave * ROWS_PER_WAVE;

    // h in WMMA C/D layout: h[t][r] = h[row = r + 8*lh][col = t*16 + ln]
    float h[4][8];
    float sumlp[8];
    int   selr[8];   // mx table row per sample-row: 0 = zero input, 1+s afterwards
#pragma unroll
    for (int t = 0; t < 4; ++t)
#pragma unroll
        for (int r = 0; r < 8; ++r) h[t][r] = 0.0f;
#pragma unroll
    for (int r = 0; r < 8; ++r) { sumlp[r] = 0.0f; selr[r] = 0; }

    union AFrag { v16h v; _Float16 e[16]; };

#pragma unroll 1
    for (int step = 0; step < NSITES; ++step) {
        // ---- 1) stage h (f32, D-layout) into per-wave LDS buffer ----
#pragma unroll
        for (int t = 0; t < 4; ++t)
#pragma unroll
            for (int r = 0; r < 8; ++r)
                hbuf[wave][(r + 8 * lh) * HPITCH + t * 16 + ln] = h[t][r];
        __syncthreads();

        // ---- 2) gather A fragments of h (16x32 f16 each, k0 = 0 and 32) ----
        AFrag a0, a1;
#pragma unroll
        for (int frag = 0; frag < 2; ++frag) {
            const int k0 = frag * 32;
#pragma unroll
            for (int part = 0; part < 2; ++part) {
                const int kb = k0 + lh * 16 + part * 8;   // 8 consecutive K values
                const float4* p = (const float4*)&hbuf[wave][ln * HPITCH + kb];
                float4 x0 = p[0], x1 = p[1];
                AFrag* a = frag ? &a1 : &a0;
                a->e[part * 8 + 0] = (_Float16)x0.x;
                a->e[part * 8 + 1] = (_Float16)x0.y;
                a->e[part * 8 + 2] = (_Float16)x0.z;
                a->e[part * 8 + 3] = (_Float16)x0.w;
                a->e[part * 8 + 4] = (_Float16)x1.x;
                a->e[part * 8 + 5] = (_Float16)x1.y;
                a->e[part * 8 + 6] = (_Float16)x1.z;
                a->e[part * 8 + 7] = (_Float16)x1.w;
            }
        }

        // ---- 3) mh = h @ Wh via 12 n-tiles x 2 k-slices of WMMA f16->f32 ----
        float acc[12][8];
#pragma unroll
        for (int nt = 0; nt < 12; ++nt) {
            const int n = nt * 16 + ln;
            v16h b0f = *(const v16h*)(whT + n * 64 +  0 + lh * 16);
            v16h b1f = *(const v16h*)(whT + n * 64 + 32 + lh * 16);
            v8f c = {0.f, 0.f, 0.f, 0.f, 0.f, 0.f, 0.f, 0.f};
            c = __builtin_amdgcn_wmma_f32_16x16x32_f16(false, a0.v, false, b0f,
                                                       (short)0, c, false, false);
            c = __builtin_amdgcn_wmma_f32_16x16x32_f16(false, a1.v, false, b1f,
                                                       (short)0, c, false, false);
#pragma unroll
            for (int r = 0; r < 8; ++r) acc[nt][r] = c[r];
        }

        // ---- 4) GRU gates (reset_after=True) ----
#pragma unroll
        for (int t = 0; t < 4; ++t) {
            const int j = t * 16 + ln;
#pragma unroll
            for (int r = 0; r < 8; ++r) {
                const int sel = selr[r];
                float mz = mxTab[sel * 192 +       j];
                float mr = mxTab[sel * 192 +  64 + j];
                float mh = mxTab[sel * 192 + 128 + j];
                float hz = acc[t][r]     + b1s[j];
                float hr = acc[4 + t][r] + b1s[64 + j];
                float hhp = acc[8 + t][r] + b1s[128 + j];
                float z  = sigmoidf_(mz + hz);
                float rg = sigmoidf_(mr + hr);
                float cand = tanhf(mh + rg * hhp);
                h[t][r] = z * h[t][r] + (1.0f - z) * cand;
            }
        }

        // ---- 5) logits = h @ Wd + bd ; log-softmax ; sample ; emit ----
#pragma unroll
        for (int r = 0; r < 8; ++r) {
            float p0 = 0.0f, p1 = 0.0f;
#pragma unroll
            for (int t = 0; t < 4; ++t) {
                const int k = t * 16 + ln;
                p0 += h[t][r] * WdL[k * 2 + 0];
                p1 += h[t][r] * WdL[k * 2 + 1];
            }
            // reduce over 16 lanes holding this row's columns (stay within half-wave)
            p0 += __shfl_xor(p0, 1, 32);  p1 += __shfl_xor(p1, 1, 32);
            p0 += __shfl_xor(p0, 2, 32);  p1 += __shfl_xor(p1, 2, 32);
            p0 += __shfl_xor(p0, 4, 32);  p1 += __shfl_xor(p1, 4, 32);
            p0 += __shfl_xor(p0, 8, 32);  p1 += __shfl_xor(p1, 8, 32);

            float l0 = p0 + bdL[0];
            float l1 = p1 + bdL[1];
            float m   = fmaxf(l0, l1);
            float lse = m + __logf(__expf(l0 - m) + __expf(l1 - m));
            float lp0 = l0 - lse, lp1 = l1 - lse;

            const int row = rowBase + r + 8 * lh;
            float u = uhash((unsigned)step, (unsigned)row);
            int s = (u < __expf(lp0)) ? 0 : 1;
            sumlp[r] += s ? lp1 : lp0;
            selr[r] = 1 + s;

            if (ln == r)  // one writer per row (lanes 0-7 / 16-23)
                out[(size_t)row * NSITES + step] = (float)s;
        }
        __syncthreads();   // hbuf reuse fence / keep waves roughly in phase
    }

    // ---- final: log_probs = 0.5 * sum ----
#pragma unroll
    for (int r = 0; r < 8; ++r) {
        if (ln == r) {
            const int row = rowBase + r + 8 * lh;
            out[(size_t)BATCH * NSITES + row] = 0.5f * sumlp[r];
        }
    }
}

extern "C" void kernel_launch(void* const* d_in, const int* in_sizes, int n_in,
                              void* d_out, int out_size, void* d_ws, size_t ws_size,
                              hipStream_t stream) {
    (void)in_sizes; (void)n_in; (void)d_ws; (void)ws_size; (void)out_size;
    // d_in[0] = numsamples (int scalar, fixed 8192 per reference)
    const float* Wi = (const float*)d_in[1];
    const float* Wh = (const float*)d_in[2];
    const float* b  = (const float*)d_in[3];
    const float* Wd = (const float*)d_in[4];
    const float* bd = (const float*)d_in[5];
    float* out = (float*)d_out;

    dim3 grid(BATCH / BLOCK_ROWS);   // 64 blocks
    dim3 block(256);                 // 8 wave32 waves, 16 samples each
    rnn_wavefn_sample_kernel<<<grid, block, 0, stream>>>(Wi, Wh, b, Wd, bd, out);
}
